// RegionProposalNetwork_33706903339041
// MI455X (gfx1250) — matrix operand order repaired and checked by reading, not compiled, not run on hardware
//
#include <hip/hip_runtime.h>
#include <hip/hip_bf16.h>

typedef __attribute__((ext_vector_type(16))) __bf16 v16bf;
typedef __attribute__((ext_vector_type(8)))  float  v8f;

#define IMGSZ   512.0f
#define BATCH   16
#define CCH     128
#define NANCH   16368
#define KTOP    4910
#define POUT    1000
#define SORT_N  16384
#define WLDS    (9 * CCH * CCH)          // 147456 bf16 weights = 288 KB

union Frag16 { v16bf v; unsigned short u[16]; };

// ---------------------------------------------------------------------------
// 3x3 conv as implicit GEMM: M = B*side*side, N = 128, K = 9*128.
// Block = 256 threads (8 waves). The FULL weight tensor is staged once per
// block into 288 KB of LDS as bf16, transposed [tap][co][ci] so a WMMA B
// fragment is one aligned 32-byte LDS read. Each block then computes 4
// groups of 128 output positions (512 x 128 outputs), one 16x128 strip per
// wave per group: one A fragment reused across 8 WMMAs.
// ---------------------------------------------------------------------------
__global__ __launch_bounds__(256)
void conv3x3_wmma(const float* __restrict__ x,     // (B, side, side, 128)
                  const float* __restrict__ w,     // (3, 3, 128, 128)
                  const float* __restrict__ bias,  // (128)
                  float* __restrict__ f,           // (B, side, side, 128)
                  int side, int Mtotal)
{
    extern __shared__ char smem[];
    __bf16* ldsw = (__bf16*)smem;        // [t][co][ci]  288 KB

    const int tid  = threadIdx.x;
    const int lane = tid & 31;
    const int wave = tid >> 5;
    const int g    = lane >> 4;          // lane group 0/1
    const int n    = lane & 15;          // A row within tile / B,C column
    const int plane = side * side;

    // ---- one-time weight staging: w[t, ci, co] -> ldsw[(t*128+co)*128+ci]
    for (int idx = tid; idx < WLDS; idx += 256) {
        const int t_  = idx >> 14;               // 16384 elements per tap
        const int rem = idx & 16383;
        const int ci  = rem >> 7;
        const int co  = rem & 127;
        ldsw[(t_ * CCH + co) * CCH + ci] = (__bf16)w[(t_ * CCH + ci) * CCH + co];
    }
    __syncthreads();

    const float bv = bias[g * 0 + 0];    // placeholder to keep bias load late
    (void)bv;

    for (int mt = 0; mt < 4; ++mt) {
        const int Mgroup = blockIdx.x * 512 + mt * 128;
        if (Mgroup >= Mtotal) continue;          // block-uniform guard
        const int Mbase = Mgroup + wave * 16;

        // this lane's A-row (output position)
        const int Mrow = Mbase + n;
        const int b    = Mrow / plane;
        const int rem  = Mrow - b * plane;
        const int oy   = rem / side;
        const int ox   = rem - oy * side;

        v8f acc[8];
#pragma unroll
        for (int q = 0; q < 8; ++q)
            acc[q] = (v8f){0.f, 0.f, 0.f, 0.f, 0.f, 0.f, 0.f, 0.f};

        for (int t = 0; t < 9; ++t) {
            const int ky = t / 3 - 1;
            const int kx = t - (t / 3) * 3 - 1;
            const int yy = oy + ky;
            const int xx = ox + kx;
            const bool inb = (yy >= 0) && (yy < side) && (xx >= 0) && (xx < side);
            const float msk = inb ? 1.0f : 0.0f;
            const int yc = yy < 0 ? 0 : (yy >= side ? side - 1 : yy);
            const int xc = xx < 0 ? 0 : (xx >= side ? side - 1 : xx);
            const float* xrow = x + (((long)b * side + yc) * side + xc) * CCH;
            const __bf16* wrow = ldsw + (long)t * CCH * CCH;

#pragma unroll
            for (int cb = 0; cb < CCH; cb += 32) {
                // ---- A fragment: 4 x b128 loads, contiguous runs per group
                const float4 r0 = *(const float4*)(xrow + cb + 16 * g);
                const float4 r1 = *(const float4*)(xrow + cb + 16 * g + 4);
                const float4 r2 = *(const float4*)(xrow + cb + 16 * g + 8);
                const float4 r3 = *(const float4*)(xrow + cb + 16 * g + 12);
                Frag16 fa;
                fa.v[0]  = (__bf16)(r0.x * msk); fa.v[1]  = (__bf16)(r0.y * msk);
                fa.v[2]  = (__bf16)(r0.z * msk); fa.v[3]  = (__bf16)(r0.w * msk);
                fa.v[4]  = (__bf16)(r1.x * msk); fa.v[5]  = (__bf16)(r1.y * msk);
                fa.v[6]  = (__bf16)(r1.z * msk); fa.v[7]  = (__bf16)(r1.w * msk);
                fa.v[8]  = (__bf16)(r2.x * msk); fa.v[9]  = (__bf16)(r2.y * msk);
                fa.v[10] = (__bf16)(r2.z * msk); fa.v[11] = (__bf16)(r2.w * msk);
                fa.v[12] = (__bf16)(r3.x * msk); fa.v[13] = (__bf16)(r3.y * msk);
                fa.v[14] = (__bf16)(r3.z * msk); fa.v[15] = (__bf16)(r3.w * msk);

                // ---- 8 WMMAs reuse the A fragment across 128 out channels
#pragma unroll
                for (int q = 0; q < 8; ++q) {
                    const int co = q * 16 + n;
                    Frag16 fb;
                    fb.v = *(const v16bf*)(wrow + co * CCH + cb + g * 16);
                    acc[q] = __builtin_amdgcn_wmma_f32_16x16x32_bf16(
                                 false, fa.v, false, fb.v, (short)0, acc[q],
                                 false, false);
                }
            }
        }

        // C/D layout: VGPR r -> tile row r + 8*g, column = co
#pragma unroll
        for (int q = 0; q < 8; ++q) {
            const int co = q * 16 + n;
            const float bq = bias[co];
#pragma unroll
            for (int r = 0; r < 8; ++r) {
                const int Mr = Mbase + r + g * 8;
                f[(long)Mr * CCH + co] = fmaxf(acc[q][r] + bq, 0.f);
            }
        }
    }
}

// ---------------------------------------------------------------------------
// 1x1 heads (conf: 128->3, bbox: 128->12) + sigmoid + anchor decode.
// One thread per (image, position).
// ---------------------------------------------------------------------------
__global__ void head_decode(const float* __restrict__ f,     // (B, side, side, 128)
                            const float* __restrict__ Wb,    // (128, 12)
                            const float* __restrict__ bb,    // (12)
                            const float* __restrict__ Wc,    // (128, 3)
                            const float* __restrict__ bc,    // (3)
                            float* __restrict__ conf,        // (B, NANCH)
                            float* __restrict__ boxes,       // (B, NANCH, 4)
                            int side, int anchOffset, float asz)
{
    const int plane = side * side;
    const int total = BATCH * plane;
    const int id = blockIdx.x * blockDim.x + threadIdx.x;
    if (id >= total) return;
    const int b   = id / plane;
    const int pos = id - b * plane;

    const float* frow = f + (long)id * CCH;
    float ac[3]  = {bc[0], bc[1], bc[2]};
    float ab[12];
#pragma unroll
    for (int j = 0; j < 12; ++j) ab[j] = bb[j];

    for (int c = 0; c < CCH; c += 4) {
        const float4 fv = *(const float4*)(frow + c);
        const float vv[4] = {fv.x, fv.y, fv.z, fv.w};
#pragma unroll
        for (int u = 0; u < 4; ++u) {
#pragma unroll
            for (int a = 0; a < 3; ++a) ac[a] += vv[u] * Wc[(c + u) * 3 + a];
#pragma unroll
            for (int j = 0; j < 12; ++j) ab[j] += vv[u] * Wb[(c + u) * 12 + j];
        }
    }

    const float step = IMGSZ / (float)side;
    const float cx = (float)(pos % side) * step;
    const float cy = (float)(pos / side) * step;
    const float scales[3] = {0.5f, 1.0f, 2.0f};

#pragma unroll
    for (int a = 0; a < 3; ++a) {
        const int k = anchOffset + pos * 3 + a;
        conf[(long)b * NANCH + k] = 1.0f / (1.0f + expf(-ac[a]));
        const float h0 = asz / scales[a];
        const float h1 = asz * scales[a];
        const float px = cx + ab[a * 4 + 0] * h0;
        const float py = cy + ab[a * 4 + 1] * h1;
        const float hx = h0 * expf(ab[a * 4 + 2]) * 0.5f;
        const float hy = h1 * expf(ab[a * 4 + 3]) * 0.5f;
        float* bo = boxes + ((long)b * NANCH + k) * 4;
        bo[0] = fminf(fmaxf(px - hx, 0.f), IMGSZ);
        bo[1] = fminf(fmaxf(py - hy, 0.f), IMGSZ);
        bo[2] = fminf(fmaxf(px + hx, 0.f), IMGSZ);
        bo[3] = fminf(fmaxf(py + hy, 0.f), IMGSZ);
    }
}

// ---------------------------------------------------------------------------
// Per-image bitonic sort (descending) of 16384 (score, idx) pairs in LDS,
// then gather top KTOP confs + boxes.  Dynamic LDS = 16384*8 bytes.
// ---------------------------------------------------------------------------
__global__ void topk_sort(const float* __restrict__ conf,
                          const float* __restrict__ boxes,
                          float* __restrict__ topConf,
                          float* __restrict__ topBoxes)
{
    extern __shared__ char smem[];
    float* key  = (float*)smem;
    int*   sidx = (int*)(key + SORT_N);
    const int b = blockIdx.x, tid = threadIdx.x, nthr = blockDim.x;

    for (int k = tid; k < SORT_N; k += nthr) {
        key[k]  = (k < NANCH) ? conf[(long)b * NANCH + k] : -1e30f;
        sidx[k] = k;
    }
    __syncthreads();

    for (int size = 2; size <= SORT_N; size <<= 1) {
        for (int stride = size >> 1; stride > 0; stride >>= 1) {
            for (int t = tid; t < SORT_N / 2; t += nthr) {
                const int i = 2 * t - (t & (stride - 1));
                const int j = i + stride;
                const bool desc = ((i & size) == 0);
                const float ki = key[i], kj = key[j];
                if (desc ? (ki < kj) : (ki > kj)) {
                    key[i] = kj; key[j] = ki;
                    const int tmp = sidx[i]; sidx[i] = sidx[j]; sidx[j] = tmp;
                }
            }
            __syncthreads();
        }
    }

    for (int k = tid; k < KTOP; k += nthr) {
        topConf[(long)b * KTOP + k] = key[k];
        const int s = sidx[k];
#pragma unroll
        for (int c = 0; c < 4; ++c)
            topBoxes[((long)b * KTOP + k) * 4 + c] =
                boxes[((long)b * NANCH + s) * 4 + c];
    }
}

// ---------------------------------------------------------------------------
// Greedy NMS per image, boxes/area/keep in dynamic LDS.
// ---------------------------------------------------------------------------
__global__ void nms_kernel(const float* __restrict__ topBoxes,
                           int* __restrict__ keep)
{
    extern __shared__ char smem[];
    float* bx   = (float*)smem;          // KTOP*4
    float* area = bx + KTOP * 4;         // KTOP
    int*   kf   = (int*)(area + KTOP);   // KTOP
    const int b = blockIdx.x, tid = threadIdx.x, nthr = blockDim.x;

    for (int k = tid; k < KTOP; k += nthr) {
        const float x1 = topBoxes[((long)b * KTOP + k) * 4 + 0];
        const float y1 = topBoxes[((long)b * KTOP + k) * 4 + 1];
        const float x2 = topBoxes[((long)b * KTOP + k) * 4 + 2];
        const float y2 = topBoxes[((long)b * KTOP + k) * 4 + 3];
        bx[k * 4 + 0] = x1; bx[k * 4 + 1] = y1;
        bx[k * 4 + 2] = x2; bx[k * 4 + 3] = y2;
        area[k] = (x2 - x1) * (y2 - y1);
        kf[k] = 1;
    }
    __syncthreads();

    for (int i = 0; i < KTOP; ++i) {
        if (kf[i]) {   // uniform read after barrier
            const float xi1 = bx[i * 4 + 0], yi1 = bx[i * 4 + 1];
            const float xi2 = bx[i * 4 + 2], yi2 = bx[i * 4 + 3];
            const float ai  = area[i];
            for (int j = i + 1 + tid; j < KTOP; j += nthr) {
                const float iw = fmaxf(fminf(bx[j * 4 + 2], xi2) - fmaxf(bx[j * 4 + 0], xi1), 0.f);
                const float ih = fmaxf(fminf(bx[j * 4 + 3], yi2) - fmaxf(bx[j * 4 + 1], yi1), 0.f);
                const float inter = iw * ih;
                const float iou = inter / (area[j] + ai - inter + 1e-9f);
                if (iou > 0.5f) kf[j] = 0;
            }
        }
        __syncthreads();
    }

    for (int k = tid; k < KTOP; k += nthr) keep[(long)b * KTOP + k] = kf[k];
}

// ---------------------------------------------------------------------------
// select_fill: mirror the reference padding logic.
// ---------------------------------------------------------------------------
__global__ void select_fill_kernel(const float* __restrict__ topConf,
                                   const float* __restrict__ topBoxes,
                                   const int* __restrict__ keep,
                                   float* __restrict__ outConf,   // (B, POUT)
                                   float* __restrict__ outBox)    // (B, POUT, 4)
{
    __shared__ int sel[POUT];
    __shared__ int s_nsel, s_sp;
    const int b = blockIdx.x, tid = threadIdx.x, nthr = blockDim.x;

    for (int j = tid; j < POUT; j += nthr) sel[j] = 0;
    __syncthreads();

    if (tid == 0) {
        int c = 0;
        for (int k = 0; k < KTOP; ++k) {
            if (keep[(long)b * KTOP + k]) {
                if (c < POUT) sel[c] = k;
                c++;
            }
        }
        const int nsel = (c < POUT) ? c : POUT;
        const int last_idx = (nsel > 0) ? sel[nsel - 1] : 0;
        int sp = KTOP - (POUT - nsel);
        if (last_idx + 1 < sp) sp = last_idx + 1;
        s_nsel = nsel;
        s_sp = sp;
    }
    __syncthreads();

    const int nsel = s_nsel, sp = s_sp;
    for (int j = tid; j < POUT; j += nthr) {
        int idx = (j < nsel) ? sel[j] : (sp + (j - nsel));
        idx = max(0, min(idx, KTOP - 1));
        outConf[(long)b * POUT + j] = topConf[(long)b * KTOP + idx];
#pragma unroll
        for (int c = 0; c < 4; ++c)
            outBox[((long)b * POUT + j) * 4 + c] =
                topBoxes[((long)b * KTOP + idx) * 4 + c];
    }
}

// ---------------------------------------------------------------------------
extern "C" void kernel_launch(void* const* d_in, const int* in_sizes, int n_in,
                              void* d_out, int out_size, void* d_ws, size_t ws_size,
                              hipStream_t stream)
{
    static const int   sides[5] = {64, 32, 16, 8, 4};
    static const float asz[5]   = {24.f, 48.f, 64.f, 156.f, 224.f};
    static const int   aoff[5]  = {0, 12288, 15360, 16128, 16320};

    const float* x[5];
    for (int l = 0; l < 5; ++l) x[l] = (const float*)d_in[l];
    const float* W_in   = (const float*)d_in[5];   // (5,3,3,128,128)
    const float* b_in   = (const float*)d_in[6];   // (5,128)
    const float* W_bbox = (const float*)d_in[7];   // (5,1,1,128,12)
    const float* b_bbox = (const float*)d_in[8];   // (5,12)
    const float* W_conf = (const float*)d_in[9];   // (5,1,1,128,3)
    const float* b_conf = (const float*)d_in[10];  // (5,3)

    // workspace layout
    const size_t featElems = (size_t)BATCH * 5456 * CCH;   // all levels concatenated
    float* feat     = (float*)d_ws;
    float* conf     = feat + featElems;                    // B*NANCH
    float* boxes    = conf + (size_t)BATCH * NANCH;        // B*NANCH*4
    float* topConf  = boxes + (size_t)BATCH * NANCH * 4;   // B*KTOP
    float* topBoxes = topConf + (size_t)BATCH * KTOP;      // B*KTOP*4
    int*   keep     = (int*)(topBoxes + (size_t)BATCH * KTOP * 4);

    // 1) convs + 2) heads per level
    size_t featOff = 0;
    for (int l = 0; l < 5; ++l) {
        const int side  = sides[l];
        const int M     = BATCH * side * side;
        float* fl = feat + featOff;

        conv3x3_wmma<<<(M + 511) / 512, 256, WLDS * sizeof(__bf16), stream>>>(
            x[l], W_in + (size_t)l * 3 * 3 * CCH * CCH, b_in + l * CCH, fl,
            side, M);

        const int threads = 256;
        head_decode<<<(M + threads - 1) / threads, threads, 0, stream>>>(
            fl, W_bbox + (size_t)l * CCH * 12, b_bbox + l * 12,
            W_conf + (size_t)l * CCH * 3, b_conf + l * 3,
            conf, boxes, side, aoff[l], asz[l]);

        featOff += (size_t)M * CCH;
    }

    // 3) top-K via bitonic sort (one workgroup per image)
    topk_sort<<<BATCH, 1024, SORT_N * (sizeof(float) + sizeof(int)), stream>>>(
        conf, boxes, topConf, topBoxes);

    // 4) NMS (one workgroup per image)
    const size_t nmsLds = (size_t)KTOP * 4 * sizeof(float) + KTOP * sizeof(float) + KTOP * sizeof(int);
    nms_kernel<<<BATCH, 1024, nmsLds, stream>>>(topBoxes, keep);

    // 5) select_fill -> outputs: conf (B,POUT) then boxes (B,POUT,4), flat
    float* outConf = (float*)d_out;
    float* outBox  = outConf + (size_t)BATCH * POUT;
    select_fill_kernel<<<BATCH, 256, 0, stream>>>(topConf, topBoxes, keep, outConf, outBox);
}
